// MultiHeadAttention_39230231281993
// MI455X (gfx1250) — matrix operand order repaired and checked by reading, not compiled, not run on hardware
//
#include <hip/hip_runtime.h>
#include <hip/hip_bf16.h>

// ---------------- problem constants ----------------
#define HEADS 16
#define DM    1024      // d_model
#define DK    64        // per-head dim (== D_V)
#define TT    2048      // sequence length
#define BB    2         // batch
#define MROWS (BB*TT)   // 4096 flattened token rows

typedef __bf16 bf16_t;
typedef __attribute__((ext_vector_type(16))) __bf16 bf16x16;
typedef __attribute__((ext_vector_type(8)))  __bf16 bf16x8;
typedef __attribute__((ext_vector_type(8)))  float  f32x8;
typedef __attribute__((ext_vector_type(4)))  int    v4i;

#define AS1 __attribute__((address_space(1)))
#define AS3 __attribute__((address_space(3)))

#define NEG_BIG (-3.0e38f)

// gfx1250 async global->LDS path (ASYNCcnt-tracked), with portable fallback
#if defined(__HIP_DEVICE_COMPILE__) && \
    __has_builtin(__builtin_amdgcn_global_load_async_to_lds_b128) && \
    __has_builtin(__builtin_amdgcn_s_wait_asynccnt)
#define HAVE_ASYNC_LDS 1
#else
#define HAVE_ASYNC_LDS 0
#endif

// copy 16 bytes global -> LDS
static __device__ inline void cp16_g2l(const bf16_t* g, bf16_t* l) {
#if HAVE_ASYNC_LDS
  // signature (from clang diagnostic): (AS1 v4i*, AS3 v4i*, imm offset, cpol)
  __builtin_amdgcn_global_load_async_to_lds_b128(
      (AS1 v4i*)g, (AS3 v4i*)l, /*offset=*/0, /*cpol=*/0);
#else
  *(bf16x8*)l = *(const bf16x8*)g;
#endif
}
static __device__ inline void wait_async_lds() {
#if HAVE_ASYNC_LDS
  __builtin_amdgcn_s_wait_asynccnt(0);
#endif
}

// D = A(16x32 bf16) * B(32x16 bf16) + C(16x16 f32)
static __device__ inline f32x8 wmma_bf16(bf16x16 a, bf16x16 b, f32x8 c) {
  return __builtin_amdgcn_wmma_f32_16x16x32_bf16(
      /*neg_a=*/false, a, /*neg_b=*/false, b,
      /*c_mod=*/(short)0, c, /*reuse_a=*/false, /*reuse_b=*/false);
}

static __device__ inline bf16x16 comb(bf16x8 lo, bf16x8 hi) {
  bf16x16 r;
#pragma unroll
  for (int i = 0; i < 8; ++i) { r[i] = lo[i]; r[8 + i] = hi[i]; }
  return r;
}

// A-fragment (16x32, bf16, row-major source, ld = row stride in elements).
// Layout: lane<16 -> row=lane, K = {0..15}; lane>=16 -> row=lane-16, K = {16..31}.
static __device__ inline bf16x16 load_afrag(const bf16_t* __restrict__ base,
                                            int row, int k0, int ld, int lane) {
  const int m   = lane & 15;
  const int kof = (lane < 16) ? 0 : 16;
  const bf16_t* p = base + (size_t)(row + m) * ld + k0 + kof;
  bf16x8 lo = *(const bf16x8*)(p);
  bf16x8 hi = *(const bf16x8*)(p + 8);
  return comb(lo, hi);
}

static __device__ inline float rmax16(float v) {
#pragma unroll
  for (int off = 1; off < 16; off <<= 1) v = fmaxf(v, __shfl_xor(v, off, 32));
  return v;
}
static __device__ inline float rsum16(float v) {
#pragma unroll
  for (int off = 1; off < 16; off <<= 1) v += __shfl_xor(v, off, 32);
  return v;
}

// ---------------- conversion kernels ----------------
__global__ __launch_bounds__(256) void k_cvt_x(const float* __restrict__ x,
                                               bf16_t* __restrict__ xb) {
  int i = blockIdx.x * 256 + threadIdx.x;
  if (i < MROWS * DM) xb[i] = (bf16_t)x[i];
}

// wt[n][k] = w[k][n]  (both 1024x1024); B-fragments then read contiguous rows of wt
__global__ __launch_bounds__(256) void k_cvt_wT(const float* __restrict__ w,
                                                bf16_t* __restrict__ wt) {
  int i = blockIdx.x * 256 + threadIdx.x;
  if (i < DM * DM) {
    int k = i >> 10, n = i & 1023;
    wt[(size_t)n * DM + k] = (bf16_t)w[(size_t)k * DM + n];
  }
}

// ---------------- fused QKV projection ----------------
// z=0: Q (scaled by 1/8)  -> qb  [b,h,t,64]
// z=1: K                  -> kb  [b,h,t,64]
// z=2: V (transposed)     -> vtb [b,h,64,t]
__global__ __launch_bounds__(128) void k_qkv_gemm(
    const bf16_t* __restrict__ xb,
    const bf16_t* __restrict__ wqT, const bf16_t* __restrict__ wkT,
    const bf16_t* __restrict__ wvT,
    bf16_t* __restrict__ qb, bf16_t* __restrict__ kb, bf16_t* __restrict__ vtb) {
  const int lane = threadIdx.x & 31;
  const int wave = threadIdx.x >> 5;
  const int row0 = blockIdx.x * 64 + wave * 16;
  const int col0 = blockIdx.y * 64;
  const int z    = blockIdx.z;
  const bf16_t* wT = (z == 0) ? wqT : (z == 1) ? wkT : wvT;

  const int ncol = lane & 15;               // B-frag / C-frag column
  const int kofB = (lane < 16) ? 0 : 16;    // B-frag K range per half-wave

  f32x8 acc[4];
#pragma unroll
  for (int nt = 0; nt < 4; ++nt)
#pragma unroll
    for (int r = 0; r < 8; ++r) acc[nt][r] = 0.0f;

  for (int k0 = 0; k0 < DM; k0 += 32) {
    __builtin_prefetch(xb + (size_t)(row0 + ncol) * DM + k0 + 64, 0, 1);
    bf16x16 a = load_afrag(xb, row0, k0, DM, lane);
#pragma unroll
    for (int nt = 0; nt < 4; ++nt) {
      const bf16_t* bp = wT + (size_t)(col0 + nt * 16 + ncol) * DM + k0 + kofB;
      bf16x16 b = *(const bf16x16*)bp;
      acc[nt] = wmma_bf16(a, b, acc[nt]);
    }
  }

  const int mb = (lane < 16) ? 0 : 8;       // C-frag row base per half-wave
#pragma unroll
  for (int nt = 0; nt < 4; ++nt) {
#pragma unroll
    for (int r = 0; r < 8; ++r) {
      int m = row0 + mb + r;                // global token row
      int n = col0 + nt * 16 + ncol;        // global feature col
      int bq = m >> 11, t = m & (TT - 1);   // T = 2048
      int h  = n >> 6,  d = n & 63;
      float v = acc[nt][r];
      size_t bh = (size_t)(bq * HEADS + h);
      if (z == 0)      qb [(bh * TT + t) * DK + d] = (bf16_t)(v * 0.125f);
      else if (z == 1) kb [(bh * TT + t) * DK + d] = (bf16_t)v;
      else             vtb[(bh * DK + d) * TT + t] = (bf16_t)v;
    }
  }
}

// ---------------- flash attention ----------------
// Block = 4 waves = 4 consecutive query tiles of the SAME (b,h).
// K/V 32-key chunks are staged once per block into double-buffered LDS via the
// gfx1250 async global->LDS path, overlapping the WMMA work on the live buffer.
__global__ __launch_bounds__(128) void k_attn(
    const bf16_t* __restrict__ qb, const bf16_t* __restrict__ kb,
    const bf16_t* __restrict__ vtb, bf16_t* __restrict__ ctx) {
  __shared__ bf16_t Kl[2][32][64];          // [buf][key][d]   8 KB
  __shared__ bf16_t Vl[2][64][32];          // [buf][d][key]   8 KB
  __shared__ bf16_t plds[4][16][32];        // per-wave P tile 4 KB

  const int tid  = threadIdx.x;
  const int lane = tid & 31;
  const int wave = tid >> 5;
  const int qg   = blockIdx.x & 31;         // query-tile group (4 tiles each)
  const int bh   = blockIdx.x >> 5;         // 0..31
  const int qt0  = qg * 4;
  const int qtile = qt0 + wave;
  const int qrow0 = qtile * 16;

  const bf16_t* Q  = qb  + (size_t)bh * TT * DK;
  const bf16_t* K  = kb  + (size_t)bh * TT * DK;
  const bf16_t* Vt = vtb + (size_t)bh * DK * TT;

  const int ncol = lane & 15;
  const int mb   = (lane < 16) ? 0 : 8;
  const int kofA = (lane < 16) ? 0 : 16;
  const int kofB = (lane < 16) ? 0 : 16;

  // Q A-fragments covering d = 0..31 and 32..63 (1/8 scale folded in)
  bf16x16 a0 = load_afrag(Q, qrow0, 0, DK, lane);
  bf16x16 a1 = load_afrag(Q, qrow0, 32, DK, lane);

  float m_i[8], l_i[8];
  f32x8 o[4];
#pragma unroll
  for (int r = 0; r < 8; ++r) { m_i[r] = NEG_BIG; l_i[r] = 0.0f; }
#pragma unroll
  for (int dt = 0; dt < 4; ++dt)
#pragma unroll
    for (int r = 0; r < 8; ++r) o[dt][r] = 0.0f;

  const int nch_w   = (qrow0 + 47) >> 5;             // this wave's causal chunks
  const int nch_max = ((qt0 + 3) * 16 + 47) >> 5;    // block-uniform trip count

  // ---- stage chunk 0 into buffer 0 (128 threads x 4 async b128 each) ----
  {
    const int kb0 = 0;
#pragma unroll
    for (int j = 0; j < 2; ++j) {
      int idx = tid + 128 * j;
      int key = idx >> 3, part = idx & 7;            // K: 32 rows x 8 x 16B
      cp16_g2l(K + (size_t)(kb0 + key) * DK + part * 8, &Kl[0][key][part * 8]);
    }
#pragma unroll
    for (int j = 0; j < 2; ++j) {
      int idx = tid + 128 * j;
      int d = idx >> 2, part = idx & 3;              // V: 64 rows x 4 x 16B
      cp16_g2l(Vt + (size_t)d * TT + kb0 + part * 8, &Vl[0][d][part * 8]);
    }
  }

  for (int c = 0; c < nch_max; ++c) {
    wait_async_lds();          // my stage of buffer (c&1) complete
    __syncthreads();           // everyone's stage complete -> buffer readable
    const int cur = c & 1, nxt = cur ^ 1;

    // prefetch next chunk into the other buffer while we compute
    if (c + 1 < nch_max) {
      const int kb1 = (c + 1) << 5;
#pragma unroll
      for (int j = 0; j < 2; ++j) {
        int idx = tid + 128 * j;
        int key = idx >> 3, part = idx & 7;
        cp16_g2l(K + (size_t)(kb1 + key) * DK + part * 8, &Kl[nxt][key][part * 8]);
      }
#pragma unroll
      for (int j = 0; j < 2; ++j) {
        int idx = tid + 128 * j;
        int d = idx >> 2, part = idx & 3;
        cp16_g2l(Vt + (size_t)d * TT + kb1 + part * 8, &Vl[nxt][d][part * 8]);
      }
    }

    if (c < nch_w) {           // causal: this wave still has work in this chunk
      const int kb0 = c << 5;

      // ---- S = Q K^T for 16 rows x 32 keys (two 16-wide C-fragments) ----
      f32x8 s0, s1;
#pragma unroll
      for (int r = 0; r < 8; ++r) { s0[r] = 0.0f; s1[r] = 0.0f; }
      {
        bf16x16 b00 = *(const bf16x16*)&Kl[cur][ncol][kofB];
        bf16x16 b01 = *(const bf16x16*)&Kl[cur][ncol][32 + kofB];
        bf16x16 b10 = *(const bf16x16*)&Kl[cur][16 + ncol][kofB];
        bf16x16 b11 = *(const bf16x16*)&Kl[cur][16 + ncol][32 + kofB];
        s0 = wmma_bf16(a0, b00, s0);
        s0 = wmma_bf16(a1, b01, s0);
        s1 = wmma_bf16(a0, b10, s1);
        s1 = wmma_bf16(a1, b11, s1);
      }

      // ---- causal mask (only the chunk straddling the diagonal) ----
      if (kb0 + 31 > qrow0 + 15) {
#pragma unroll
        for (int r = 0; r < 8; ++r) {
          int row = qrow0 + mb + r;
          if (kb0 + ncol      > row) s0[r] = NEG_BIG;
          if (kb0 + 16 + ncol > row) s1[r] = NEG_BIG;
        }
      }

      // ---- online softmax (row stats across each 16-lane half) ----
      float cf[8];
#pragma unroll
      for (int r = 0; r < 8; ++r) {
        float mx   = rmax16(fmaxf(s0[r], s1[r]));
        float mnew = fmaxf(m_i[r], mx);
        float e0 = __expf(s0[r] - mnew);
        float e1 = __expf(s1[r] - mnew);
        s0[r] = e0; s1[r] = e1;
        float sum = rsum16(e0 + e1);
        cf[r]  = __expf(m_i[r] - mnew);
        l_i[r] = l_i[r] * cf[r] + sum;
        m_i[r] = mnew;
      }
#pragma unroll
      for (int dt = 0; dt < 4; ++dt)
#pragma unroll
        for (int r = 0; r < 8; ++r) o[dt][r] *= cf[r];

      // ---- P: C-layout -> A-fragment layout via per-wave LDS tile ----
#pragma unroll
      for (int r = 0; r < 8; ++r) {
        plds[wave][mb + r][ncol]      = (bf16_t)s0[r];
        plds[wave][mb + r][16 + ncol] = (bf16_t)s1[r];
      }
      bf16x8 plo = *(const bf16x8*)&plds[wave][ncol][kofA];
      bf16x8 phi = *(const bf16x8*)&plds[wave][ncol][kofA + 8];
      bf16x16 pa = comb(plo, phi);

      // ---- O += P * V  (V staged [d][key] so B-frag is a contiguous row) ----
#pragma unroll
      for (int dt = 0; dt < 4; ++dt) {
        bf16x16 vb = *(const bf16x16*)&Vl[cur][dt * 16 + ncol][kofB];
        o[dt] = wmma_bf16(pa, vb, o[dt]);
      }
    }

    __syncthreads();           // all waves done reading `cur` before restage
  }

  // ---- normalize and write ctx as bf16 [token, h*64+d] for the Wo GEMM ----
  const int bq = bh >> 4, h = bh & 15;
  float inv[8];
#pragma unroll
  for (int r = 0; r < 8; ++r) inv[r] = 1.0f / l_i[r];
#pragma unroll
  for (int dt = 0; dt < 4; ++dt)
#pragma unroll
    for (int r = 0; r < 8; ++r) {
      size_t m = (size_t)bq * TT + qrow0 + mb + r;
      ctx[m * DM + h * DK + dt * 16 + ncol] = (bf16_t)(o[dt][r] * inv[r]);
    }
}

// ---------------- output projection: out = ctx @ Wo (f32 result) ----------
__global__ __launch_bounds__(128) void k_out_gemm(
    const bf16_t* __restrict__ ctx, const bf16_t* __restrict__ woT,
    float* __restrict__ out) {
  const int lane = threadIdx.x & 31;
  const int wave = threadIdx.x >> 5;
  const int row0 = blockIdx.x * 64 + wave * 16;
  const int col0 = blockIdx.y * 64;

  const int ncol = lane & 15;
  const int kofB = (lane < 16) ? 0 : 16;

  f32x8 acc[4];
#pragma unroll
  for (int nt = 0; nt < 4; ++nt)
#pragma unroll
    for (int r = 0; r < 8; ++r) acc[nt][r] = 0.0f;

  for (int k0 = 0; k0 < DM; k0 += 32) {
    __builtin_prefetch(ctx + (size_t)(row0 + ncol) * DM + k0 + 64, 0, 1);
    bf16x16 a = load_afrag(ctx, row0, k0, DM, lane);
#pragma unroll
    for (int nt = 0; nt < 4; ++nt) {
      const bf16_t* bp = woT + (size_t)(col0 + nt * 16 + ncol) * DM + k0 + kofB;
      bf16x16 b = *(const bf16x16*)bp;
      acc[nt] = wmma_bf16(a, b, acc[nt]);
    }
  }

  const int mb = (lane < 16) ? 0 : 8;
#pragma unroll
  for (int nt = 0; nt < 4; ++nt)
#pragma unroll
    for (int r = 0; r < 8; ++r) {
      size_t m = (size_t)(row0 + mb + r);
      out[m * DM + col0 + nt * 16 + ncol] = acc[nt][r];
    }
}

// ---------------- host-side launch ----------------
extern "C" void kernel_launch(void* const* d_in, const int* in_sizes, int n_in,
                              void* d_out, int out_size, void* d_ws, size_t ws_size,
                              hipStream_t stream) {
  (void)in_sizes; (void)n_in; (void)out_size; (void)ws_size;
  const float* x  = (const float*)d_in[0];
  const float* Wq = (const float*)d_in[1];
  const float* Wk = (const float*)d_in[2];
  const float* Wv = (const float*)d_in[3];
  const float* Wo = (const float*)d_in[4];
  float* out = (float*)d_out;

  char* p = (char*)d_ws;
  auto carve = [&](size_t nelem) -> bf16_t* {
    bf16_t* r = (bf16_t*)p;
    p += (nelem * sizeof(bf16_t) + 255) & ~(size_t)255;
    return r;
  };
  bf16_t* xb  = carve((size_t)MROWS * DM);
  bf16_t* wqT = carve((size_t)DM * DM);
  bf16_t* wkT = carve((size_t)DM * DM);
  bf16_t* wvT = carve((size_t)DM * DM);
  bf16_t* woT = carve((size_t)DM * DM);
  bf16_t* qb  = carve((size_t)BB * HEADS * TT * DK);
  bf16_t* kb  = carve((size_t)BB * HEADS * TT * DK);
  bf16_t* vtb = carve((size_t)BB * HEADS * DK * TT);
  bf16_t* ctx = carve((size_t)MROWS * DM);

  k_cvt_x<<<(MROWS * DM + 255) / 256, 256, 0, stream>>>(x, xb);
  k_cvt_wT<<<(DM * DM + 255) / 256, 256, 0, stream>>>(Wq, wqT);
  k_cvt_wT<<<(DM * DM + 255) / 256, 256, 0, stream>>>(Wk, wkT);
  k_cvt_wT<<<(DM * DM + 255) / 256, 256, 0, stream>>>(Wv, wvT);
  k_cvt_wT<<<(DM * DM + 255) / 256, 256, 0, stream>>>(Wo, woT);

  dim3 gq(MROWS / 64, DM / 64, 3);
  k_qkv_gemm<<<gq, 128, 0, stream>>>(xb, wqT, wkT, wvT, qb, kb, vtb);

  k_attn<<<32 * 32, 128, 0, stream>>>(qb, kb, vtb, ctx);

  dim3 go(MROWS / 64, DM / 64);
  k_out_gemm<<<go, 128, 0, stream>>>(ctx, woT, out);
}
